// ScreeningAttentionTriton_9345848836722
// MI455X (gfx1250) — matrix operand order repaired and checked by reading, not compiled, not run on hardware
//
#include <hip/hip_runtime.h>
#include <hip/hip_bf16.h>

// Problem constants (B=2, T=2048, D_MODEL=768, H=12, D=64)
#define BATCH 2
#define TSEQ  2048
#define DMODEL 768
#define NHEAD 12
#define HDIM  64
#define NTOK  (BATCH * TSEQ)                 // 4096
#define BHTD  (BATCH * NHEAD * TSEQ * HDIM)  // 3145728 f16 elems

typedef __attribute__((ext_vector_type(16))) _Float16 v16h;
typedef __attribute__((ext_vector_type(2)))  _Float16 h2;
typedef __attribute__((ext_vector_type(8)))  float    v8f;

#define WMMA_F16(a, b, c) \
  __builtin_amdgcn_wmma_f32_16x16x32_f16(false, (a), false, (b), (short)0, (c), false, false)

// ---------------------------------------------------------------------------
// Fragment-ready LDS layout. Each WMMA operand is stored as [lane][16 halves]
// so the read side is one contiguous 32B v16h load per fragment.
//
// A 16x32 f16 (ISA): lane = (m&15) + 16*g with g=(k>>3)&1; within lane,
//   slot = 2j + (k&1), j = k'<8 ? k'/2 : 4+(k'-16)/2, k' = k-8g.
// B 32x16 f16 (ISA): lane = (n&15) + 16*g, slot as for A (n in place of m).
// C 16x16 f32 (ISA): lane -> n = lane&15, VGPR r -> m = r + 8*(lane>>4).
// ---------------------------------------------------------------------------
static __device__ __forceinline__ int a_lane_slot(int k31, int m15, int* slot) {
  int g  = (k31 >> 3) & 1;
  int k2 = k31 - 8 * g;                       // in {0..7} or {16..23}
  int j  = (k2 < 8) ? (k2 >> 1) : (4 + ((k2 - 16) >> 1));
  *slot  = 2 * j + (k31 & 1);
  return m15 + 16 * g;
}

static __device__ __forceinline__ v8f zero8() {
  v8f z;
#pragma unroll
  for (int r = 0; r < 8; ++r) z[r] = 0.0f;
  return z;
}

static __device__ __forceinline__
void st_c(float* d, int stride, int m0, int n0, int lane, v8f c) {
  int n = lane & 15, g = lane >> 4;
#pragma unroll
  for (int r = 0; r < 8; ++r)
    d[(m0 + r + 8 * g) * stride + n0 + n] = c[r];
}

// ---------------------------------------------------------------------------
// Kernel 1: P = X @ W^T (4096x768 * 768x768), per-head L2 normalize,
// store f16 into [B][H][T][D]. Block = 64 tokens x 1 head.
// ---------------------------------------------------------------------------
__global__ __launch_bounds__(256)
void proj_norm_kernel(const float* __restrict__ X, const float* __restrict__ W,
                      _Float16* __restrict__ out) {
  __shared__ __align__(32) _Float16 fragA[4][32][16];  // [msub][lane][slot]
  __shared__ __align__(32) _Float16 fragB[4][32][16];  // [nsub][lane][slot], A mapping with n as m
  __shared__ float Pt[64][65];
  __shared__ float red[64][4];

  const int tid  = threadIdx.x;
  const int lane = tid & 31;
  const int wv   = tid >> 5;       // 0..7
  const int wm   = wv >> 1;        // msub 0..3
  const int wn   = wv & 1;         // 0..1
  const int m0   = wm * 16;

  const int tile_m = blockIdx.x;
  const int head   = blockIdx.y;
  const int jbase  = head * HDIM;

  v8f acc[2];
  acc[0] = zero8(); acc[1] = zero8();

  for (int kk = 0; kk < DMODEL; kk += 32) {
    if (kk + 32 < DMODEL) {   // prefetch next K-slice (global_prefetch_b8)
      __builtin_prefetch(&X[(size_t)(tile_m * 64 + (tid >> 2)) * DMODEL + kk + 32], 0, 1);
      __builtin_prefetch(&W[(size_t)(jbase + (tid >> 2)) * DMODEL + kk + 32], 0, 1);
    }
    // A tile: X[64 x 32] -> fragment-ready layout, packed pair stores
    for (int e = tid; e < 1024; e += 256) {
      int kp = e & 15, m = e >> 4, k = kp * 2;
      float x0 = X[(size_t)(tile_m * 64 + m) * DMODEL + kk + k];
      float x1 = X[(size_t)(tile_m * 64 + m) * DMODEL + kk + k + 1];
      int slot; int la = a_lane_slot(k, m & 15, &slot);
      h2 v; v[0] = (_Float16)x0; v[1] = (_Float16)x1;
      *(h2*)&fragA[m >> 4][la][slot] = v;
    }
    // B tile: B[k][n] = W[jbase+n][kk+k] -> same lane/slot mapping as A (n in place of m)
    for (int e = tid; e < 1024; e += 256) {
      int kp = e & 15, n = e >> 4, k = kp * 2;
      float w0 = W[(size_t)(jbase + n) * DMODEL + kk + k];
      float w1 = W[(size_t)(jbase + n) * DMODEL + kk + k + 1];
      h2 v; v[0] = (_Float16)w0; v[1] = (_Float16)w1;
      int slot; int lb = a_lane_slot(k, n & 15, &slot);
      *(h2*)&fragB[n >> 4][lb][slot] = v;
    }
    __syncthreads();

    v16h a = *(const v16h*)&fragA[wm][lane][0];
#pragma unroll
    for (int ns = 0; ns < 2; ++ns) {
      v16h b = *(const v16h*)&fragB[wn * 2 + ns][lane][0];
      acc[ns] = WMMA_F16(a, b, acc[ns]);
    }
    __syncthreads();
  }

  st_c(&Pt[0][0], 65, m0, wn * 32, lane, acc[0]);
  st_c(&Pt[0][0], 65, m0, wn * 32 + 16, lane, acc[1]);
  __syncthreads();

  // row-wise L2 normalize over 64 cols: x / max(||x||, 1e-8)
  {
    int row = tid >> 2, q = tid & 3;
    float ss = 0.f;
#pragma unroll
    for (int c = 0; c < 16; ++c) {
      float v = Pt[row][q * 16 + c];
      ss += v * v;
    }
    red[row][q] = ss;
    __syncthreads();
    float n = sqrtf(red[row][0] + red[row][1] + red[row][2] + red[row][3]);
    float scale = 1.0f / fmaxf(n, 1e-8f);
    int gi = tile_m * 64 + row;
    int b  = gi / TSEQ, t = gi % TSEQ;
    _Float16* dst = out + (((size_t)(b * NHEAD + head) * TSEQ + t) * HDIM);
#pragma unroll
    for (int c = 0; c < 16; ++c) {
      int d = q * 16 + c;
      dst[d] = (_Float16)(Pt[row][d] * scale);
    }
  }
}

// ---------------------------------------------------------------------------
// Kernel 2: causal screening attention per (b,h). Block = 64-query tile.
//   S = Q K^T;  alpha = r * clip((S+1)/2,0,1)^w  (causal);  O = alpha @ V
//   TanhNorm rows; store f16 [B][T][DMODEL].
// ---------------------------------------------------------------------------
__global__ __launch_bounds__(256)
void screen_attn_kernel(const _Float16* __restrict__ qh,
                        const _Float16* __restrict__ kh,
                        const _Float16* __restrict__ vh,
                        const float* __restrict__ s_v,
                        const float* __restrict__ s_r,
                        _Float16* __restrict__ oh) {
  __shared__ __align__(32) _Float16 fQ[4][2][32][16];  // A: [msub][dsub][lane][slot]
  __shared__ __align__(32) _Float16 fK[2][4][32][16];  // B for S: [dsub][keysub][lane][slot], A mapping with key as m
  __shared__ __align__(32) _Float16 fV[2][4][32][16];  // B for O: [keysub][dsub][lane][slot], A mapping with d as m
  __shared__ __align__(32) _Float16 fS[4][2][32][16];  // A for O: [msub][keysub][lane][slot]
  __shared__ float Ot[64][65];
  __shared__ float red[64][4];

  const int tid  = threadIdx.x;
  const int lane = tid & 31;
  const int wv   = tid >> 5;
  const int wm   = wv >> 1;        // msub
  const int wn   = wv & 1;
  const int m0   = wm * 16;

  const int qt = blockIdx.x;               // query tile 0..31
  const int bh = blockIdx.y;               // 0..23
  const int b  = bh / NHEAD, h = bh % NHEAD;

  const float w_h = expf(s_v[h]) + 1.0f;
  const float r_h = expf(s_r[h]) + 1.0f;

  const _Float16* qbase = qh + (size_t)bh * TSEQ * HDIM;
  const _Float16* kbase = kh + (size_t)bh * TSEQ * HDIM;
  const _Float16* vbase = vh + (size_t)bh * TSEQ * HDIM;

  // Q tile -> A-fragment layout (written once)
  for (int e = tid; e < 2048; e += 256) {
    int dp = e & 31, m = e >> 5, d = dp * 2;
    h2 v = *(const h2*)&qbase[(size_t)(qt * 64 + m) * HDIM + d];
    int slot; int la = a_lane_slot(d & 31, m & 15, &slot);
    *(h2*)&fQ[m >> 4][d >> 5][la][slot] = v;
  }

  v8f o[2];
  o[0] = zero8(); o[1] = zero8();

  for (int kt = 0; kt <= qt; ++kt) {
    // K^T as B operand: B[d][key] -> same lane/slot mapping as A (key in place of m)
    for (int e = tid; e < 2048; e += 256) {
      int dp = e & 31, key = e >> 5, d = dp * 2;
      h2 v = *(const h2*)&kbase[(size_t)(kt * 64 + key) * HDIM + d];
      int slot; int lb = a_lane_slot(d & 31, key & 15, &slot);
      *(h2*)&fK[d >> 5][key >> 4][lb][slot] = v;
    }
    // V as B operand: B[key][d] -> same lane/slot mapping as A (d in place of m)
    for (int e = tid; e < 2048; e += 256) {
      int d = e & 63, kp = e >> 6, key = kp * 2;
      h2 v;
      v[0] = vbase[(size_t)(kt * 64 + key) * HDIM + d];
      v[1] = vbase[(size_t)(kt * 64 + key + 1) * HDIM + d];
      int slot; int lb = a_lane_slot(key & 31, d & 15, &slot);
      *(h2*)&fV[key >> 5][d >> 4][lb][slot] = v;
    }
    __syncthreads();

    // S = Q K^T, elementwise screening, write alpha straight into A-frag layout
#pragma unroll
    for (int ns = 0; ns < 2; ++ns) {
      int nsub = wn * 2 + ns;
      int n0   = nsub * 16;
      v8f s = zero8();
#pragma unroll
      for (int dsub = 0; dsub < 2; ++dsub) {
        v16h a = *(const v16h*)&fQ[wm][dsub][lane][0];
        v16h bb = *(const v16h*)&fK[dsub][nsub][lane][0];
        s = WMMA_F16(a, bb, s);
      }
      int nloc  = n0 + (lane & 15);          // key column this lane holds
      int kglob = kt * 64 + nloc;
      int gc    = lane >> 4;
      int ksub  = nloc >> 5;
      int slot;
      int ga_lane_base = a_lane_slot(nloc & 31, 0, &slot); // = 16*g_a
#pragma unroll
      for (int r = 0; r < 8; ++r) {
        int mloc  = m0 + r + 8 * gc;
        int mglob = qt * 64 + mloc;
        float base = fminf(fmaxf((s[r] + 1.0f) * 0.5f, 0.0f), 1.0f);
        float al = (base > 0.0f && kglob <= mglob)
                     ? r_h * exp2f(w_h * log2f(base)) : 0.0f;
        fS[wm][ksub][(r + 8 * gc) + ga_lane_base][slot] = (_Float16)al;
      }
    }
    __syncthreads();

    // O += alpha @ V
#pragma unroll
    for (int ns = 0; ns < 2; ++ns) {
      int nsub = wn * 2 + ns;
#pragma unroll
      for (int ksub = 0; ksub < 2; ++ksub) {
        v16h a = *(const v16h*)&fS[wm][ksub][lane][0];
        v16h bb = *(const v16h*)&fV[ksub][nsub][lane][0];
        o[ns] = WMMA_F16(a, bb, o[ns]);
      }
    }
    __syncthreads();
  }

  st_c(&Ot[0][0], 65, m0, wn * 32, lane, o[0]);
  st_c(&Ot[0][0], 65, m0, wn * 32 + 16, lane, o[1]);
  __syncthreads();

  // TanhNorm: out = tanh(||o||) * o / (||o|| + eps); store [B][T][DMODEL]
  {
    int row = tid >> 2, q = tid & 3;
    float ss = 0.f;
#pragma unroll
    for (int c = 0; c < 16; ++c) {
      float v = Ot[row][q * 16 + c];
      ss += v * v;
    }
    red[row][q] = ss;
    __syncthreads();
    float n = sqrtf(red[row][0] + red[row][1] + red[row][2] + red[row][3]);
    float scale = tanhf(n) / (n + 1e-8f);
    int t = qt * 64 + row;
    _Float16* dst = oh + ((size_t)(b * TSEQ + t) * DMODEL + h * HDIM);
#pragma unroll
    for (int c = 0; c < 16; ++c) {
      int d = q * 16 + c;
      dst[d] = (_Float16)(Ot[row][d] * scale);
    }
  }
}

// ---------------------------------------------------------------------------
// Kernel 3: out = Oh @ Wo^T, f32 written straight from C fragments.
// ---------------------------------------------------------------------------
__global__ __launch_bounds__(256)
void out_gemm_kernel(const _Float16* __restrict__ Oh, const float* __restrict__ Wo,
                     float* __restrict__ out) {
  __shared__ __align__(32) _Float16 fragA[4][32][16];
  __shared__ __align__(32) _Float16 fragB[4][32][16];

  const int tid  = threadIdx.x;
  const int lane = tid & 31;
  const int wv   = tid >> 5;
  const int wm   = wv >> 1;
  const int wn   = wv & 1;
  const int m0   = wm * 16;

  const int tile_m = blockIdx.x;
  const int jbase  = blockIdx.y * 64;

  v8f acc[2];
  acc[0] = zero8(); acc[1] = zero8();

  for (int kk = 0; kk < DMODEL; kk += 32) {
    if (kk + 32 < DMODEL) {
      __builtin_prefetch(&Oh[(size_t)(tile_m * 64 + (tid >> 2)) * DMODEL + kk + 32], 0, 1);
      __builtin_prefetch(&Wo[(size_t)(jbase + (tid >> 2)) * DMODEL + kk + 32], 0, 1);
    }
    for (int e = tid; e < 1024; e += 256) {
      int kp = e & 15, m = e >> 4, k = kp * 2;
      h2 v = *(const h2*)&Oh[(size_t)(tile_m * 64 + m) * DMODEL + kk + k];
      int slot; int la = a_lane_slot(k, m & 15, &slot);
      *(h2*)&fragA[m >> 4][la][slot] = v;
    }
    for (int e = tid; e < 1024; e += 256) {
      int kp = e & 15, n = e >> 4, k = kp * 2;
      float w0 = Wo[(size_t)(jbase + n) * DMODEL + kk + k];
      float w1 = Wo[(size_t)(jbase + n) * DMODEL + kk + k + 1];
      h2 v; v[0] = (_Float16)w0; v[1] = (_Float16)w1;
      int slot; int lb = a_lane_slot(k, n & 15, &slot);
      *(h2*)&fragB[n >> 4][lb][slot] = v;
    }
    __syncthreads();

    v16h a = *(const v16h*)&fragA[wm][lane][0];
#pragma unroll
    for (int ns = 0; ns < 2; ++ns) {
      v16h bt = *(const v16h*)&fragB[wn * 2 + ns][lane][0];
      acc[ns] = WMMA_F16(a, bt, acc[ns]);
    }
    __syncthreads();
  }

  int n = lane & 15, g = lane >> 4;
#pragma unroll
  for (int ns = 0; ns < 2; ++ns) {
    int ncol = jbase + wn * 32 + ns * 16 + n;
#pragma unroll
    for (int r = 0; r < 8; ++r) {
      int mrow = tile_m * 64 + m0 + r + 8 * g;
      out[(size_t)mrow * DMODEL + ncol] = acc[ns][r];
    }
  }
}

// ---------------------------------------------------------------------------
extern "C" void kernel_launch(void* const* d_in, const int* in_sizes, int n_in,
                              void* d_out, int out_size, void* d_ws, size_t ws_size,
                              hipStream_t stream) {
  (void)in_sizes; (void)n_in; (void)out_size; (void)ws_size;
  const float* query = (const float*)d_in[0];
  const float* key   = (const float*)d_in[1];
  const float* value = (const float*)d_in[2];
  const float* Wq    = (const float*)d_in[3];
  const float* Wk    = (const float*)d_in[4];
  const float* Wv    = (const float*)d_in[5];
  const float* Wo    = (const float*)d_in[6];
  const float* s_v   = (const float*)d_in[7];
  const float* s_r   = (const float*)d_in[8];

  _Float16* qh = (_Float16*)d_ws;                       // [B][H][T][D]
  _Float16* kh = qh + (size_t)BHTD;
  _Float16* vh = kh + (size_t)BHTD;
  _Float16* oh = vh + (size_t)BHTD;                     // [B][T][DMODEL]

  dim3 blk(256);
  dim3 gproj(NTOK / 64, NHEAD);
  proj_norm_kernel<<<gproj, blk, 0, stream>>>(query, Wq, qh);
  proj_norm_kernel<<<gproj, blk, 0, stream>>>(key,   Wk, kh);
  proj_norm_kernel<<<gproj, blk, 0, stream>>>(value, Wv, vh);

  dim3 gattn(TSEQ / 64, BATCH * NHEAD);
  screen_attn_kernel<<<gattn, blk, 0, stream>>>(qh, kh, vh, s_v, s_r, oh);

  dim3 gout(NTOK / 64, DMODEL / 64);
  out_gemm_kernel<<<gout, blk, 0, stream>>>(oh, Wo, (float*)d_out);
}
